// HierarchicalAttention_54640573940155
// MI455X (gfx1250) — matrix operand order repaired and checked by reading, not compiled, not run on hardware
//
#include <hip/hip_runtime.h>
#include <hip/hip_bf16.h>
#include <math.h>

#define NB 32
#define NT 2048
#define ND 1024
#define NA 128

typedef __attribute__((ext_vector_type(16))) __bf16 v16bf;
typedef __attribute__((ext_vector_type(8)))  float  v8f;
typedef __attribute__((ext_vector_type(4)))  float  v4f;

// ---------------------------------------------------------------------------
// Kernel 0: repack Ws (f32 [D,A] row-major) into bf16 WMMA B-operand layout.
// For tile (kb, nt), lane l holds 16 bf16: element j = Ws[kb*32 + 8*(l>>4) + j + (j>=8 ? 8 : 0)][nt*16 + (l&15)]
// (B 32x16 16-bit layout: lanes 0-15 carry K=0..7,16..23 of column N=lane, lanes 16-31 carry K=8..15,24..31)
// ---------------------------------------------------------------------------
__global__ void k_pack_ws(const float* __restrict__ Ws, __bf16* __restrict__ Wp) {
  int idx  = blockIdx.x * blockDim.x + threadIdx.x;   // 0..8191  (32 kb * 8 nt * 32 lanes)
  int lane = idx & 31;
  int nt   = (idx >> 5) & 7;
  int kb   = idx >> 8;
  int col  = nt * 16 + (lane & 15);
  int kbase = kb * 32 + ((lane >> 4) ? 8 : 0);
  v16bf v;
#pragma unroll
  for (int j = 0; j < 16; ++j)
    v[j] = (__bf16)Ws[(kbase + j + (j >= 8 ? 8 : 0)) * NA + col];
  *(v16bf*)(Wp + (size_t)idx * 16) = v;
}

// ---------------------------------------------------------------------------
// Kernel 1: scores[b,t] = sum_a tanh((x @ Ws)[a] + bs[a]) * us[a]
// 8 waves/block, each wave owns one 16-row M-tile, accumulates all 8 N-tiles
// of A=128 via v_wmma_f32_16x16x32_bf16 over K=1024 (32 steps).
// ---------------------------------------------------------------------------
__global__ void k_scores(const float* __restrict__ x, const __bf16* __restrict__ Wp,
                         const float* __restrict__ bs, const float* __restrict__ us,
                         float* __restrict__ scores) {
  const int lane = threadIdx.x & 31;
  const int wave = threadIdx.x >> 5;                 // 0..7
  const int lcol = lane & 15;
  const int half = lane >> 4;                        // 0 or 1
  const int rowBase = (blockIdx.x * 8 + wave) * 16;  // 16-row tile of B*T=65536
  const float* xrow = x + (size_t)(rowBase + lcol) * ND;
  const int koff = half * 8;                         // A-matrix 16-bit layout K offset

  v8f acc[8] = {};                                   // 8 N-tiles x 8 VGPRs f32

  for (int kb = 0; kb < 32; ++kb) {
    // A tile 16x32 bf16: lane holds row=lcol, K in {koff..koff+7, 16+koff..16+koff+7}
    const float* ap = xrow + kb * 32 + koff;
    v4f a0 = *(const v4f*)(ap);
    v4f a1 = *(const v4f*)(ap + 4);
    v4f a2 = *(const v4f*)(ap + 16);
    v4f a3 = *(const v4f*)(ap + 20);
    v16bf Amat;
#pragma unroll
    for (int j = 0; j < 4; ++j) {
      Amat[j]      = (__bf16)a0[j];
      Amat[4 + j]  = (__bf16)a1[j];
      Amat[8 + j]  = (__bf16)a2[j];
      Amat[12 + j] = (__bf16)a3[j];
    }
    const v16bf* bp = (const v16bf*)Wp + ((size_t)kb * 8) * 32 + lane;
#pragma unroll
    for (int nt = 0; nt < 8; ++nt) {
      v16bf Bmat = bp[nt * 32];                      // pre-packed B tile, L2-hot
      acc[nt] = __builtin_amdgcn_wmma_f32_16x16x32_bf16(
          false, Amat, false, Bmat, (short)0, acc[nt], false, false);
    }
  }

  // Epilogue: tanh + dot with us, fused. C/D layout: element j -> row M=j+8*half, col N=lcol(+16*nt)
  float p[8] = {0.f, 0.f, 0.f, 0.f, 0.f, 0.f, 0.f, 0.f};
#pragma unroll
  for (int nt = 0; nt < 8; ++nt) {
    int col = nt * 16 + lcol;
    float bsv = bs[col];
    float usv = us[col];
#pragma unroll
    for (int j = 0; j < 8; ++j)
      p[j] += tanhf(acc[nt][j] + bsv) * usv;
  }
  // reduce across the 16 column-lanes (stays within each half: xor bits 0..3)
#pragma unroll
  for (int j = 0; j < 8; ++j) {
    float v = p[j];
    v += __shfl_xor(v, 8, 32);
    v += __shfl_xor(v, 4, 32);
    v += __shfl_xor(v, 2, 32);
    v += __shfl_xor(v, 1, 32);
    p[j] = v;
  }
  if (lcol == 0) {
#pragma unroll
    for (int j = 0; j < 8; ++j)
      scores[rowBase + half * 8 + j] = p[j];
  }
}

// ---------------------------------------------------------------------------
// Kernel 2: per-batch softmax over T -> weights; also zero out[] for kernel 3.
// ---------------------------------------------------------------------------
__global__ void k_softmax(const float* __restrict__ scores, float* __restrict__ weights,
                          float* __restrict__ out) {
  __shared__ float sdata[256];
  const int b = blockIdx.x;
  const int tid = threadIdx.x;
  const float* s = scores + (size_t)b * NT;

  float m = -INFINITY;
  for (int t = tid; t < NT; t += 256) m = fmaxf(m, s[t]);
  sdata[tid] = m;
  __syncthreads();
  for (int off = 128; off > 0; off >>= 1) {
    if (tid < off) sdata[tid] = fmaxf(sdata[tid], sdata[tid + off]);
    __syncthreads();
  }
  const float M = sdata[0];
  __syncthreads();

  float e = 0.f;
  for (int t = tid; t < NT; t += 256) e += __expf(s[t] - M);
  sdata[tid] = e;
  __syncthreads();
  for (int off = 128; off > 0; off >>= 1) {
    if (tid < off) sdata[tid] += sdata[tid + off];
    __syncthreads();
  }
  const float inv = 1.0f / sdata[0];

  for (int t = tid; t < NT; t += 256)
    weights[(size_t)b * NT + t] = __expf(s[t] - M) * inv;
  for (int d = tid; d < ND; d += 256)
    out[(size_t)b * ND + d] = 0.f;
}

// ---------------------------------------------------------------------------
// Kernel 3: out[b,d] += sum_t weights[b,t] * x[b,t,d]; grid (B, D/256, T/256).
// ---------------------------------------------------------------------------
__global__ void k_wsum(const float* __restrict__ x, const float* __restrict__ weights,
                       float* __restrict__ out) {
  const int b  = blockIdx.x;
  const int d  = blockIdx.y * 256 + threadIdx.x;
  const int t0 = blockIdx.z * 256;
  const float* w  = weights + (size_t)b * NT + t0;
  const float* xp = x + ((size_t)b * NT + t0) * ND + d;
  const bool leader = (threadIdx.x & 31) == 0;

  float acc = 0.f;
#pragma unroll 4
  for (int t = 0; t < 256; ++t) {
    if (leader && t < 248)
      __builtin_prefetch(xp + (size_t)(t + 8) * ND, 0, 0);  // global_prefetch_b8
    acc += w[t] * xp[(size_t)t * ND];
  }
  atomicAdd(&out[(size_t)b * ND + d], acc);
}

// ---------------------------------------------------------------------------
extern "C" void kernel_launch(void* const* d_in, const int* in_sizes, int n_in,
                              void* d_out, int out_size, void* d_ws, size_t ws_size,
                              hipStream_t stream) {
  const float* x  = (const float*)d_in[0];   // [B,T,D]
  const float* Ws = (const float*)d_in[1];   // [D,A]
  const float* bs = (const float*)d_in[2];   // [A]
  const float* us = (const float*)d_in[3];   // [A]
  float* out = (float*)d_out;                // [B,D]

  char* wsb = (char*)d_ws;
  __bf16* Wp     = (__bf16*)(wsb);                  // 256 KB packed bf16 Ws
  float*  scores = (float*)(wsb + (256 << 10));     // 256 KB [B,T]
  float*  weight = (float*)(wsb + (512 << 10));     // 256 KB [B,T]

  k_pack_ws<<<32, 256, 0, stream>>>(Ws, Wp);
  k_scores<<<512, 256, 0, stream>>>(x, Wp, bs, us, scores);
  k_softmax<<<NB, 256, 0, stream>>>(scores, weight, out);
  k_wsum<<<dim3(NB, ND / 256, NT / 256), 256, 0, stream>>>(x, weight, out);
}
